// Experiment6_670014898913
// MI455X (gfx1250) — matrix operand order repaired and checked by reading, not compiled, not run on hardware
//
#include <hip/hip_runtime.h>
#include <math.h>

// ---------------- model constants ----------------
#define BB    8
#define LL    512
#define DM    512
#define DF    2048
#define HH    4
#define DH    128
#define PRED  96
#define DS    16
#define DTR   32
#define ELAY  2

#define ACT_NONE     0
#define ACT_RELU     1
#define ACT_SOFTPLUS 2

typedef __attribute__((ext_vector_type(16))) __bf16 v16bf;
typedef __attribute__((ext_vector_type(8)))  float  v8f;

union BF16x16 { v16bf v; __bf16 h[16]; };

__device__ __forceinline__ float silu_f(float v) {
  return v / (1.0f + __expf(-v));
}

// ---------------- generic batched WMMA GEMM (templated, LDS-staged B) ----------------
// C[m,n] = act( sum_k A[m,k] * (TRANSB ? W[n,k] : W[k,n]) + bias[n] )
// block: 256 threads = 8 waves; each wave: one 16-row tile x 32 cols (2 WMMA acc).
// grid: (N/32, M/128, nbatch); requires M%128==0, N%32==0, K%32==0 (all calls satisfy).
// per-batch offsets: off = (z/batch_h)*s + (z%batch_h)*s2
template<int TRANSB, int ACT>
__global__ void wmma_gemm(const float* __restrict__ A, const float* __restrict__ W,
                          const float* __restrict__ bias, float* __restrict__ C,
                          int M, int N, int K, int lda, int ldw, int ldc,
                          long long sA, long long sA2, long long sW, long long sW2,
                          long long sC, long long sC2, int batch_h)
{
  int z  = blockIdx.z;
  int zb = z / batch_h, zh = z % batch_h;
  A += zb * sA + zh * sA2;
  W += zb * sW + zh * sW2;
  C += zb * sC + zh * sC2;

  __shared__ __bf16 Bs[32][32];          // [n-local][k-local], 2 KB

  int tid  = threadIdx.x;
  int lane = tid & 31;
  int wave = tid >> 5;
  int tm   = blockIdx.y * 8 + wave;      // 16-row tile index
  int tn2  = blockIdx.x;                 // 32-col tile index
  int half = lane >> 4;
  int nloc = lane & 15;
  int mrow = tm * 16 + nloc;             // A row for this lane
  int ncol0 = tn2 * 32 + nloc;
  int ncol1 = ncol0 + 16;

  // B-fill coordinates: one float4 per thread covers the whole 32x32 tile
  int fb_n, fb_k;
  if (TRANSB) { fb_n = tid >> 3;  fb_k = (tid & 7) * 4; }   // contiguous in k
  else        { fb_k = tid >> 3;  fb_n = (tid & 7) * 4; }   // contiguous in n

  v8f acc0 = {0.f,0.f,0.f,0.f,0.f,0.f,0.f,0.f};
  v8f acc1 = {0.f,0.f,0.f,0.f,0.f,0.f,0.f,0.f};

  const float* Arow = A + (long long)mrow * lda;

  for (int kb = 0; kb < K; kb += 32) {
    // ---- cooperative B tile -> bf16 LDS (one b128 global load per thread) ----
    if (TRANSB) {
      const float4 w = *(const float4*)&W[(long long)(tn2 * 32 + fb_n) * ldw + (kb + fb_k)];
      Bs[fb_n][fb_k + 0] = (__bf16)w.x;
      Bs[fb_n][fb_k + 1] = (__bf16)w.y;
      Bs[fb_n][fb_k + 2] = (__bf16)w.z;
      Bs[fb_n][fb_k + 3] = (__bf16)w.w;
    } else {
      const float4 w = *(const float4*)&W[(long long)(kb + fb_k) * ldw + (tn2 * 32 + fb_n)];
      Bs[fb_n + 0][fb_k] = (__bf16)w.x;
      Bs[fb_n + 1][fb_k] = (__bf16)w.y;
      Bs[fb_n + 2][fb_k] = (__bf16)w.z;
      Bs[fb_n + 3][fb_k] = (__bf16)w.w;
    }
    __syncthreads();

    // ---- A fragment: one contiguous 16-float run per lane (4x b128 loads) ----
    // A 16x32 layout (matches B): K = half*16 + e
    const float4 a0 = *(const float4*)&Arow[kb + half * 16];
    const float4 a1 = *(const float4*)&Arow[kb + half * 16 + 4];
    const float4 a2 = *(const float4*)&Arow[kb + half * 16 + 8];
    const float4 a3 = *(const float4*)&Arow[kb + half * 16 + 12];
    float af[16] = { a0.x, a0.y, a0.z, a0.w,  a1.x, a1.y, a1.z, a1.w,
                     a2.x, a2.y, a2.z, a2.w,  a3.x, a3.y, a3.z, a3.w };
    BF16x16 a;
#pragma unroll
    for (int e = 0; e < 16; ++e) a.h[e] = (__bf16)af[e];

    // ---- B fragments from LDS: K = half*16 + e, contiguous 32B per lane ----
    v16bf b0 = *(const v16bf*)&Bs[nloc][half * 16];
    v16bf b1 = *(const v16bf*)&Bs[nloc + 16][half * 16];

    acc0 = __builtin_amdgcn_wmma_f32_16x16x32_bf16(false, a.v, false, b0,
                                                   (short)0, acc0, false, false);
    acc1 = __builtin_amdgcn_wmma_f32_16x16x32_bf16(false, a.v, false, b1,
                                                   (short)0, acc1, false, false);
    __syncthreads();
  }

  float bv0 = bias ? bias[ncol0] : 0.0f;
  float bv1 = bias ? bias[ncol1] : 0.0f;
#pragma unroll
  for (int r = 0; r < 8; ++r) {
    int m = tm * 16 + r + 8 * half;
    float v0 = acc0[r] + bv0;
    float v1 = acc1[r] + bv1;
    if (ACT == ACT_RELU) {
      v0 = fmaxf(v0, 0.0f); v1 = fmaxf(v1, 0.0f);
    } else if (ACT == ACT_SOFTPLUS) {
      v0 = (v0 > 20.0f) ? v0 : log1pf(__expf(v0));
      v1 = (v1 > 20.0f) ? v1 : log1pf(__expf(v1));
    }
    C[(long long)m * ldc + ncol0] = v0;
    C[(long long)m * ldc + ncol1] = v1;
  }
}

// ---------------- RevIN stats ----------------
__global__ void stats_kernel(const float* __restrict__ x, float* __restrict__ stats)
{
  int b = blockIdx.x;
  __shared__ float sh[4][256];
  float s0 = 0.f, s1 = 0.f, q0 = 0.f, q1 = 0.f;
  for (int l = threadIdx.x; l < LL; l += 256) {
    float v0 = x[((long long)b * LL + l) * 2 + 0];
    float v1 = x[((long long)b * LL + l) * 2 + 1];
    s0 += v0; s1 += v1; q0 += v0 * v0; q1 += v1 * v1;
  }
  sh[0][threadIdx.x] = s0; sh[1][threadIdx.x] = s1;
  sh[2][threadIdx.x] = q0; sh[3][threadIdx.x] = q1;
  __syncthreads();
  for (int off = 128; off > 0; off >>= 1) {
    if (threadIdx.x < off)
      for (int j = 0; j < 4; ++j)
        sh[j][threadIdx.x] += sh[j][threadIdx.x + off];
    __syncthreads();
  }
  if (threadIdx.x == 0) {
    float m0 = sh[0][0] / LL, m1 = sh[1][0] / LL;
    float v0 = sh[2][0] / LL - m0 * m0;
    float v1 = sh[3][0] / LL - m1 * m1;
    stats[b * 4 + 0] = m0;
    stats[b * 4 + 1] = m1;
    stats[b * 4 + 2] = sqrtf(v0 + 1e-5f);
    stats[b * 4 + 3] = sqrtf(v1 + 1e-5f);
  }
}

// ---------------- embedding: pp = xn @ Wp + bp ----------------
__global__ void embed_kernel(const float* __restrict__ x, const float* __restrict__ stats,
                             const float* __restrict__ Wp, const float* __restrict__ bp,
                             float* __restrict__ pp)
{
  long long row = blockIdx.x;           // b*L + l
  int b = (int)(row / LL);
  float xn0 = (x[row * 2 + 0] - stats[b * 4 + 0]) / stats[b * 4 + 2];
  float xn1 = (x[row * 2 + 1] - stats[b * 4 + 1]) / stats[b * 4 + 3];
  for (int d = threadIdx.x; d < DM; d += blockDim.x)
    pp[row * DM + d] = xn0 * Wp[d] + xn1 * Wp[DM + d] + bp[d];
}

__global__ void bsum_kernel(const float* a, const float* b, float* o)
{
  int i = threadIdx.x + blockIdx.x * blockDim.x;
  if (i < DM) o[i] = a[i] + b[i];
}

// ---------------- row softmax (in place) ----------------
__global__ void softmax_kernel(float* __restrict__ S, int width, float scale)
{
  long long row = blockIdx.x;
  float* p = S + row * (long long)width;
  __shared__ float red[256];
  float mx = -1e30f;
  for (int i = threadIdx.x; i < width; i += 256) mx = fmaxf(mx, p[i] * scale);
  red[threadIdx.x] = mx; __syncthreads();
  for (int off = 128; off > 0; off >>= 1) {
    if (threadIdx.x < off) red[threadIdx.x] = fmaxf(red[threadIdx.x], red[threadIdx.x + off]);
    __syncthreads();
  }
  mx = red[0]; __syncthreads();
  float sum = 0.f;
  for (int i = threadIdx.x; i < width; i += 256) {
    float e = __expf(p[i] * scale - mx);
    p[i] = e; sum += e;
  }
  red[threadIdx.x] = sum; __syncthreads();
  for (int off = 128; off > 0; off >>= 1) {
    if (threadIdx.x < off) red[threadIdx.x] += red[threadIdx.x + off];
    __syncthreads();
  }
  float inv = 1.0f / red[0];
  for (int i = threadIdx.x; i < width; i += 256) p[i] *= inv;
}

// ---------------- depthwise causal conv (width 2) + silu ----------------
__global__ void conv_silu_kernel(const float* __restrict__ xz, const float* __restrict__ cw,
                                 const float* __restrict__ cb, float* __restrict__ xc)
{
  long long row = blockIdx.x;           // b*L + l
  int l = (int)(row % LL);
  for (int d = threadIdx.x; d < DM; d += 256) {
    float x1 = xz[row * (2 * DM) + d];
    float x0 = (l > 0) ? xz[(row - 1) * (2 * DM) + d] : 0.0f;
    float v  = cw[d * 2 + 0] * x0 + cw[d * 2 + 1] * x1 + cb[d];
    xc[row * DM + d] = silu_f(v);
  }
}

// ---------------- selective scan: thread per (b,d), states in registers ----------------
__global__ void scan_kernel(const float* __restrict__ dt,  const float* __restrict__ dbl,
                            const float* __restrict__ xc,  const float* __restrict__ xz,
                            const float* __restrict__ Alog,const float* __restrict__ Dp,
                            float* __restrict__ y)
{
  int idx = blockIdx.x * blockDim.x + threadIdx.x;
  if (idx >= BB * DM) return;
  int b = idx / DM, d = idx % DM;
  float Aa[DS], hs[DS];
#pragma unroll
  for (int n = 0; n < DS; ++n) { Aa[n] = -__expf(Alog[d * DS + n]); hs[n] = 0.f; }
  float Dd = Dp[d];
  for (int l = 0; l < LL; ++l) {
    long long row = (long long)b * LL + l;
    float dtv = dt[row * DM + d];
    float xv  = xc[row * DM + d];
    float zv  = xz[row * (2 * DM) + DM + d];
    const float* bl = dbl + row * 64;
    float yv = 0.f;
#pragma unroll
    for (int n = 0; n < DS; ++n) {
      float dA = __expf(dtv * Aa[n]);
      hs[n] = dA * hs[n] + dtv * bl[DTR + n] * xv;
      yv += hs[n] * bl[DTR + DS + n];
    }
    yv += Dd * xv;
    y[row * DM + d] = yv * silu_f(zv);
  }
}

// ---------------- reverse along L ----------------
__global__ void reverse_kernel(const float* __restrict__ h, float* __restrict__ hr)
{
  long long i = (long long)blockIdx.x * blockDim.x + threadIdx.x;
  long long total = (long long)BB * LL * DM;
  if (i >= total) return;
  int d = (int)(i % DM);
  long long row = i / DM;
  int b = (int)(row / LL), l = (int)(row % LL);
  long long ri = ((long long)b * LL + (LL - 1 - l)) * DM + d;
  hr[i] = h[ri];
}

// h = h + f + reverse(r)
__global__ void combine_kernel(float* __restrict__ h, const float* __restrict__ f,
                               const float* __restrict__ r)
{
  long long i = (long long)blockIdx.x * blockDim.x + threadIdx.x;
  long long total = (long long)BB * LL * DM;
  if (i >= total) return;
  int d = (int)(i % DM);
  long long row = i / DM;
  int b = (int)(row / LL), l = (int)(row % LL);
  long long ri = ((long long)b * LL + (LL - 1 - l)) * DM + d;
  h[i] = h[i] + f[i] + r[ri];
}

// ---------------- layernorm over DM (optional residual add), in/out may alias ----------------
__global__ void ln_kernel(float* __restrict__ out, const float* __restrict__ x,
                          const float* __restrict__ addx,
                          const float* __restrict__ g, const float* __restrict__ bta)
{
  long long row = blockIdx.x;
  int t = threadIdx.x;                        // 256 threads, 2 elements each
  float v0 = x[row * DM + t]       + (addx ? addx[row * DM + t]       : 0.f);
  float v1 = x[row * DM + t + 256] + (addx ? addx[row * DM + t + 256] : 0.f);
  __shared__ float ssum[256], ssq[256];
  ssum[t] = v0 + v1; ssq[t] = v0 * v0 + v1 * v1;
  __syncthreads();
  for (int off = 128; off > 0; off >>= 1) {
    if (t < off) { ssum[t] += ssum[t + off]; ssq[t] += ssq[t + off]; }
    __syncthreads();
  }
  float mean = ssum[0] / DM;
  float var  = ssq[0] / DM - mean * mean;
  float rs   = rsqrtf(var + 1e-5f);
  out[row * DM + t]       = (v0 - mean) * rs * g[t]       + bta[t];
  out[row * DM + t + 256] = (v1 - mean) * rs * g[t + 256] + bta[t + 256];
}

// ---------------- final projection: only time steps l=0,1 matter ----------------
__global__ void proj_out_kernel(const float* __restrict__ h, const float* __restrict__ stats,
                                const float* __restrict__ pW, const float* __restrict__ pb,
                                float* __restrict__ out)
{
  int p = blockIdx.x, c = blockIdx.y, b = blockIdx.z;
  float s = 0.f;
  for (int d = threadIdx.x; d < DM; d += 64)
    s += h[((long long)b * LL + c) * DM + d] * pW[(long long)d * PRED + p];
  __shared__ float red[64];
  red[threadIdx.x] = s; __syncthreads();
  for (int off = 32; off > 0; off >>= 1) {
    if (threadIdx.x < off) red[threadIdx.x] += red[threadIdx.x + off];
    __syncthreads();
  }
  if (threadIdx.x == 0) {
    float v = red[0] + pb[p];
    out[((long long)b * PRED + p) * 2 + c] = v * stats[b * 4 + 2 + c] + stats[b * 4 + c];
  }
}

// ---------------- host side ----------------
static inline void launch_gemm(hipStream_t st, const float* A, const float* W,
                               const float* bias, float* C,
                               int M, int N, int K, int lda, int ldw, int ldc,
                               long long sA, long long sA2, long long sW, long long sW2,
                               long long sC, long long sC2,
                               int nbatch, int batch_h, int transB, int act)
{
  dim3 g(N / 32, M / 128, nbatch);
  if (transB) {
    wmma_gemm<1, ACT_NONE><<<g, 256, 0, st>>>(A, W, bias, C, M, N, K, lda, ldw, ldc,
                                              sA, sA2, sW, sW2, sC, sC2, batch_h);
  } else if (act == ACT_RELU) {
    wmma_gemm<0, ACT_RELU><<<g, 256, 0, st>>>(A, W, bias, C, M, N, K, lda, ldw, ldc,
                                              sA, sA2, sW, sW2, sC, sC2, batch_h);
  } else if (act == ACT_SOFTPLUS) {
    wmma_gemm<0, ACT_SOFTPLUS><<<g, 256, 0, st>>>(A, W, bias, C, M, N, K, lda, ldw, ldc,
                                                  sA, sA2, sW, sW2, sC, sC2, batch_h);
  } else {
    wmma_gemm<0, ACT_NONE><<<g, 256, 0, st>>>(A, W, bias, C, M, N, K, lda, ldw, ldc,
                                              sA, sA2, sW, sW2, sC, sC2, batch_h);
  }
}

extern "C" void kernel_launch(void* const* d_in, const int* in_sizes, int n_in,
                              void* d_out, int out_size, void* d_ws, size_t ws_size,
                              hipStream_t stream)
{
  const float* x_enc  = (const float*)d_in[0];
  const float* Wp     = (const float*)d_in[4];
  const float* bp     = (const float*)d_in[5];
  const float* bi     = (const float*)d_in[7];
  const float* Wq     = (const float*)d_in[8];
  const float* Wk     = (const float*)d_in[9];
  const float* Wv     = (const float*)d_in[10];
  const float* Wo     = (const float*)d_in[11];
  const float* bq     = (const float*)d_in[12];
  const float* bk     = (const float*)d_in[13];
  const float* bv     = (const float*)d_in[14];
  const float* bo     = (const float*)d_in[15];
  const float* mWin   = (const float*)d_in[16];
  const float* mConvw = (const float*)d_in[17];
  const float* mConvb = (const float*)d_in[18];
  const float* mWx    = (const float*)d_in[19];
  const float* mWdt   = (const float*)d_in[20];
  const float* mBdt   = (const float*)d_in[21];
  const float* mAlog  = (const float*)d_in[22];
  const float* mD     = (const float*)d_in[23];
  const float* mWout  = (const float*)d_in[24];
  const float* ffW1   = (const float*)d_in[25];
  const float* ffB1   = (const float*)d_in[26];
  const float* ffW2   = (const float*)d_in[27];
  const float* ffB2   = (const float*)d_in[28];
  const float* n1g    = (const float*)d_in[29];
  const float* n1b    = (const float*)d_in[30];
  const float* n2g    = (const float*)d_in[31];
  const float* n2b    = (const float*)d_in[32];
  const float* nfg    = (const float*)d_in[33];
  const float* nfb    = (const float*)d_in[34];
  const float* projW  = (const float*)d_in[35];
  const float* projB  = (const float*)d_in[36];
  float* out = (float*)d_out;
  float* ws  = (float*)d_ws;

  const long long ROWS = (long long)BB * LL;            // 4096
  const long long NTOK = ROWS * DM;                     // 2,097,152

  // workspace layout (floats)
  size_t o = 0;
  float* STATS = ws + o; o += 64;
  float* Hbuf  = ws + o; o += NTOK;
  float* T1    = ws + o; o += NTOK;
  float* T2    = ws + o; o += NTOK;
  float* T3    = ws + o; o += NTOK;
  float* SBUF  = ws + o; o += (long long)BB * HH * LL * LL;  // also FFN mid (same size)
  float* XZ    = ws + o; o += ROWS * (2 * DM);
  float* DBL   = ws + o; o += ROWS * 64;
  float* DT    = ws + o; o += NTOK;
  float* BF    = ws + o; o += NTOK;
  float* BR    = ws + o; o += NTOK;
  float* HR    = ws + o; o += NTOK;
  float* BSUM  = ws + o; o += DM;

  // ---- RevIN + embedding ----
  stats_kernel<<<BB, 256, 0, stream>>>(x_enc, STATS);
  embed_kernel<<<(int)ROWS, 256, 0, stream>>>(x_enc, STATS, Wp, bp, Hbuf);
  bsum_kernel<<<2, 256, 0, stream>>>(bo, bi, BSUM);

  // ---- MHA ----
  launch_gemm(stream, Hbuf, Wq, bq, T1, (int)ROWS, DM, DM, DM, DM, DM, 0,0,0,0,0,0, 1,1,0, ACT_NONE);
  launch_gemm(stream, Hbuf, Wk, bk, T2, (int)ROWS, DM, DM, DM, DM, DM, 0,0,0,0,0,0, 1,1,0, ACT_NONE);
  launch_gemm(stream, Hbuf, Wv, bv, T3, (int)ROWS, DM, DM, DM, DM, DM, 0,0,0,0,0,0, 1,1,0, ACT_NONE);

  const long long LDm = (long long)LL * DM;      // per-batch activation stride
  const long long LLL = (long long)LL * LL;      // per-head score stride
  // S = Q @ K^T  (batched over B*H, head slice via second offset)
  launch_gemm(stream, T1, T2, nullptr, SBUF, LL, LL, DH, DM, DM, LL,
              LDm, DH, LDm, DH, (long long)HH * LLL, LLL, BB * HH, HH, 1, ACT_NONE);
  softmax_kernel<<<(int)((long long)BB * HH * LL), 256, 0, stream>>>(SBUF, LL, 1.0f / sqrtf((float)DH));
  // O = P @ V
  launch_gemm(stream, SBUF, T3, nullptr, T1, LL, DH, LL, LL, DM, DM,
              (long long)HH * LLL, LLL, LDm, DH, LDm, DH, BB * HH, HH, 0, ACT_NONE);
  // h = O @ Wo + (bo + bi)
  launch_gemm(stream, T1, Wo, BSUM, Hbuf, (int)ROWS, DM, DM, DM, DM, DM, 0,0,0,0,0,0, 1,1,0, ACT_NONE);

  // ---- encoder layers ----
  int blocksTok = (int)((NTOK + 255) / 256);
  for (int li = 0; li < ELAY; ++li) {
    for (int dir = 0; dir < 2; ++dir) {
      const float* in = (dir == 0) ? Hbuf : HR;
      float* outb = (dir == 0) ? BF : BR;
      if (dir == 1) reverse_kernel<<<blocksTok, 256, 0, stream>>>(Hbuf, HR);
      int s = li * 2 + dir;
      launch_gemm(stream, in, mWin + (long long)s * DM * 2 * DM, nullptr, XZ,
                  (int)ROWS, 2 * DM, DM, DM, 2 * DM, 2 * DM, 0,0,0,0,0,0, 1,1,0, ACT_NONE);
      conv_silu_kernel<<<(int)ROWS, 256, 0, stream>>>(XZ, mConvw + (long long)s * DM * 2,
                                                      mConvb + (long long)s * DM, T1);
      launch_gemm(stream, T1, mWx + (long long)s * DM * 64, nullptr, DBL,
                  (int)ROWS, 64, DM, DM, 64, 64, 0,0,0,0,0,0, 1,1,0, ACT_NONE);
      launch_gemm(stream, DBL, mWdt + (long long)s * DTR * DM, mBdt + (long long)s * DM, DT,
                  (int)ROWS, DM, DTR, 64, DM, DM, 0,0,0,0,0,0, 1,1,0, ACT_SOFTPLUS);
      scan_kernel<<<(BB * DM + 255) / 256, 256, 0, stream>>>(DT, DBL, T1, XZ,
                  mAlog + (long long)s * DM * DS, mD + (long long)s * DM, T3);
      launch_gemm(stream, T3, mWout + (long long)s * DM * DM, nullptr, outb,
                  (int)ROWS, DM, DM, DM, DM, DM, 0,0,0,0,0,0, 1,1,0, ACT_NONE);
    }
    combine_kernel<<<blocksTok, 256, 0, stream>>>(Hbuf, BF, BR);
    ln_kernel<<<(int)ROWS, 256, 0, stream>>>(Hbuf, Hbuf, nullptr,
                                             n1g + (long long)li * DM, n1b + (long long)li * DM);
    // FFN
    launch_gemm(stream, Hbuf, ffW1 + (long long)li * DM * DF, ffB1 + (long long)li * DF, SBUF,
                (int)ROWS, DF, DM, DM, DF, DF, 0,0,0,0,0,0, 1,1,0, ACT_RELU);
    launch_gemm(stream, SBUF, ffW2 + (long long)li * DF * DM, ffB2 + (long long)li * DM, T1,
                (int)ROWS, DM, DF, DF, DM, DM, 0,0,0,0,0,0, 1,1,0, ACT_NONE);
    ln_kernel<<<(int)ROWS, 256, 0, stream>>>(Hbuf, Hbuf, T1,
                                             n2g + (long long)li * DM, n2b + (long long)li * DM);
  }

  // ---- final norm + projection (only l=0,1 feed the output) ----
  ln_kernel<<<(int)ROWS, 256, 0, stream>>>(Hbuf, Hbuf, nullptr, nfg, nfb);
  proj_out_kernel<<<dim3(PRED, 2, BB), 64, 0, stream>>>(Hbuf, STATS, projW, projB, out);
}
